// ContextualizedNN_2396591751282
// MI455X (gfx1250) — matrix operand order repaired and checked by reading, not compiled, not run on hardware
//
#include <hip/hip_runtime.h>
#include <hip/hip_bf16.h>

// ---------------------------------------------------------------------------
// ContextualizedNN for MI455X (gfx1250). One wave32 = one batch element.
// All matmuls computed TRANSPOSED (Xt = Et*St, H1t = W1t*Xt, H2t = W2t*H1t):
//   - A fragments: row-major LDS, 2 aligned ds_load_b128
//   - B fragments: row-major LDS, 2 aligned ds_load_b128 (or vector global f32)
//   - C fragments: 4 x v_cvt_pk_bf16_f32 + 1 ds_store_b128
//   - bias folded into the WMMA accumulator init (D = A*B + C)
// ---------------------------------------------------------------------------

typedef __attribute__((ext_vector_type(16))) __bf16 v16bf;
typedef __attribute__((ext_vector_type(2)))  __bf16 v2bf;
typedef __attribute__((ext_vector_type(8)))  float  v8f;

union Frag {
    v16bf        v;
    unsigned int u[8];
};

static __device__ __forceinline__ unsigned pk_bf16(float lo, float hi) {
    v2bf t;
    t[0] = (__bf16)lo;
    t[1] = (__bf16)hi;
    return __builtin_bit_cast(unsigned, t);     // -> v_cvt_pk_bf16_f32
}

static __device__ __forceinline__ v8f wmma_bf16(v16bf a, v16bf b, v8f c) {
    return __builtin_amdgcn_wmma_f32_16x16x32_bf16(false, a, false, b,
                                                   (short)0, c, false, false);
}

// two 16-byte chunks -> one 8-dword fragment
static __device__ __forceinline__ Frag load2x16(const char* p0, const char* p1) {
    Frag f;
    uint4 a = *(const uint4*)p0;
    uint4 b = *(const uint4*)p1;
    f.u[0] = a.x; f.u[1] = a.y; f.u[2] = a.z; f.u[3] = a.w;
    f.u[4] = b.x; f.u[5] = b.y; f.u[6] = b.z; f.u[7] = b.w;
    return f;
}
static __device__ __forceinline__ Frag load_af(const char* rowbase, int hi) {
    return load2x16(rowbase + hi * 32, rowbase + hi * 32 + 16);
}
static __device__ __forceinline__ Frag load_bf(const char* p) {
    return load2x16(p, p + 16);
}

static __device__ __forceinline__ void store_c_bf16(void* p, v8f c) {
    uint4 o;
    o.x = pk_bf16(c[0], c[1]); o.y = pk_bf16(c[2], c[3]);
    o.z = pk_bf16(c[4], c[5]); o.w = pk_bf16(c[6], c[7]);
    *(uint4*)p = o;
}
static __device__ __forceinline__ void store_c_relu_bf16(void* p, v8f c) {
    uint4 o;
    o.x = pk_bf16(fmaxf(c[0], 0.f), fmaxf(c[1], 0.f));
    o.y = pk_bf16(fmaxf(c[2], 0.f), fmaxf(c[3], 0.f));
    o.z = pk_bf16(fmaxf(c[4], 0.f), fmaxf(c[5], 0.f));
    o.w = pk_bf16(fmaxf(c[6], 0.f), fmaxf(c[7], 0.f));
    *(uint4*)p = o;
}

#define KNB   20
#define DDIM  64
#define WPB   2
#define NTHR  (WPB * 32)
#define SLAB  12288          // per-wave: ET/H1 4KB @0, X/H2 8KB @4096

#define W1T_BYTES (64 * 128 * 2)
#define W2T_BYTES (32 * 64 * 2)
#define SMEM_TOTAL (WPB * SLAB + W1T_BYTES + W2T_BYTES + 256 + 128 + 128 + 16)

__global__ __launch_bounds__(NTHR)
void cnn_fused_wmma_kernel(
    const int*   __restrict__ user_idxs,
    const int*   __restrict__ item_idxs,
    const int*   __restrict__ user_idx_tensor,
    const float* __restrict__ user_scr_tensor,
    const int*   __restrict__ item_idx_tensor,
    const float* __restrict__ item_scr_tensor,
    const float* __restrict__ user_emb_table,
    const float* __restrict__ item_emb_table,
    const float* __restrict__ w1, const float* __restrict__ b1,
    const float* __restrict__ w2, const float* __restrict__ b2,
    const float* __restrict__ w3, const float* __restrict__ b3,
    float* __restrict__ out, int B)
{
    __shared__ __align__(16) unsigned char smem[SMEM_TOTAL];

    __bf16* W1T = (__bf16*)(smem + WPB * SLAB);         // [64 out][128 k]
    __bf16* W2T = (__bf16*)((char*)W1T + W1T_BYTES);    // [32 out][64 k]
    float* sB1 = (float*)((char*)W2T + W2T_BYTES);      // 64
    float* sB2 = (float*)((char*)sB1 + 256);            // 32
    float* sW3 = (float*)((char*)sB2 + 128);            // 32
    float* sB3 = (float*)((char*)sW3 + 128);            // 1

    const int tid  = threadIdx.x;
    const int lane = tid & 31;
    const int wave = tid >> 5;

    // ---- stage weights once per block (bf16, out-major == A-matrix rows) ---
    for (int i = tid; i < 128 * 64; i += NTHR) {
        int k = i >> 6, n = i & 63;                 // w1 stored (in=128,out=64)
        W1T[n * 128 + k] = (__bf16)w1[i];
    }
    for (int i = tid; i < 64 * 32; i += NTHR) {
        int k = i >> 5, n = i & 31;                 // w2 stored (in=64,out=32)
        W2T[n * 64 + k] = (__bf16)w2[i];
    }
    if (tid < 64) sB1[tid] = b1[tid];
    if (tid < 32) { sB2[tid] = b2[tid]; sW3[tid] = w3[tid]; }
    if (tid == 0) sB3[0] = b3[0];
    __syncthreads();

    // ---- per-wave private LDS slab ----------------------------------------
    __bf16* ET = (__bf16*)(smem + wave * SLAB);     // [64 d][32 j], later H1 [32 m][64 k]
    __bf16* H1 = ET;
    __bf16* X  = (__bf16*)((char*)ET + 4096);       // [32 m][128 k], later H2 f32 [32 m][32 k]
    float*  H2 = (float*)X;

    const int gwave  = blockIdx.x * WPB + wave;
    const int nwaves = gridDim.x * WPB;

    const int nloc = lane & 15;
    const int hi   = lane >> 4;                     // 0/1: which 16-lane half

    for (int b = gwave; b < B; b += nwaves) {
        // ============ einsums as Xt = Et[64x32j] * St[32j x 32m] ===========
        for (int side = 0; side < 2; ++side) {
            const int*   idxt = side ? item_idx_tensor : user_idx_tensor;
            const float* scr  = side ? item_scr_tensor : user_scr_tensor;
            const float* emb  = side ? item_emb_table  : user_emb_table;
            const int    root = side ? item_idxs[b]    : user_idxs[b];

            int njl = (lane < KNB) ? idxt[root * KNB + lane] : 0;

            // zero K-pad columns (j = 20..31) of ET
            #pragma unroll
            for (int t = 0; t < 12; ++t) {
                int idx = t * 32 + lane;
                int d = idx / 6, j = 20 + 2 * (idx % 6);
                *(unsigned*)&ET[d * 32 + j] = 0u;
            }
            // stage E transposed, packed pairs along j
            for (int j = 0; j < KNB; j += 2) {
                int nj0 = __shfl(njl, j);
                int nj1 = __shfl(njl, j + 1);
                const float* r0 = emb + (long)nj0 * DDIM;
                const float* r1 = emb + (long)nj1 * DDIM;
                *(unsigned*)&ET[lane * 32 + j]        = pk_bf16(r0[lane],      r1[lane]);
                *(unsigned*)&ET[(lane + 32) * 32 + j] = pk_bf16(r0[lane + 32], r1[lane + 32]);
            }

            // B fragments of St: column m = batch row; score rows are 80 B,
            // 16B aligned, so j-pairs come in as vector float4 loads.
            Frag sb[2];
            #pragma unroll
            for (int nt = 0; nt < 2; ++nt) {
                int m  = nt * 16 + nloc;
                int nm = __shfl(njl, m & 31);
                const float* srow = scr + (long)nm * KNB;
                float4 q0, q1, q2, q3;
                if (hi == 0) {              // j = 0..15, all in-bounds
                    q0 = *(const float4*)(srow + 0);
                    q1 = *(const float4*)(srow + 4);
                    q2 = *(const float4*)(srow + 8);
                    q3 = *(const float4*)(srow + 12);
                } else {                    // j = 16..31, only 16..19 valid
                    q0 = *(const float4*)(srow + 16);
                    q1 = make_float4(0.f, 0.f, 0.f, 0.f);
                    q2 = q1; q3 = q1;
                }
                sb[nt].u[0] = pk_bf16(q0.x, q0.y);
                sb[nt].u[1] = pk_bf16(q0.z, q0.w);
                sb[nt].u[2] = pk_bf16(q1.x, q1.y);
                sb[nt].u[3] = pk_bf16(q1.z, q1.w);
                sb[nt].u[4] = pk_bf16(q2.x, q2.y);
                sb[nt].u[5] = pk_bf16(q2.z, q2.w);
                sb[nt].u[6] = pk_bf16(q3.x, q3.y);
                sb[nt].u[7] = pk_bf16(q3.z, q3.w);
                if (nt == 1) {              // rows m = 20..31 are zero padding
                    unsigned msk = (m < KNB) ? 0xFFFFFFFFu : 0u;
                    #pragma unroll
                    for (int i = 0; i < 8; ++i) sb[nt].u[i] &= msk;
                }
            }

            #pragma unroll
            for (int dt = 0; dt < 4; ++dt) {
                Frag a = load_af((const char*)ET + (dt * 16 + nloc) * 64, hi);
                #pragma unroll
                for (int nt = 0; nt < 2; ++nt) {
                    v8f c = {};
                    c = wmma_bf16(a.v, sb[nt].v, c);
                    int m = nt * 16 + nloc;
                    store_c_bf16((char*)X + m * 256 + (side * 64 + dt * 16 + hi * 8) * 2, c);
                }
            }
        }

        // ============ layer 1: H1t = W1t[64x128] * Xt[128x32] (+b1, ReLU) ==
        Frag bx[2][4];
        #pragma unroll
        for (int nt = 0; nt < 2; ++nt)
            #pragma unroll
            for (int ks = 0; ks < 4; ++ks)
                bx[nt][ks] = load_bf((const char*)X + (nt * 16 + nloc) * 256 + ks * 64 + hi * 32);

        #pragma unroll
        for (int ot = 0; ot < 4; ++ot) {
            v8f c0, c1;
            #pragma unroll
            for (int r = 0; r < 8; ++r) {
                float bv = sB1[ot * 16 + hi * 8 + r];
                c0[r] = bv; c1[r] = bv;
            }
            #pragma unroll
            for (int ks = 0; ks < 4; ++ks) {
                Frag a = load_af((const char*)W1T + (ot * 16 + nloc) * 256 + ks * 64, hi);
                c0 = wmma_bf16(a.v, bx[0][ks].v, c0);
                c1 = wmma_bf16(a.v, bx[1][ks].v, c1);
            }
            store_c_relu_bf16((char*)H1 + (0 * 16 + nloc) * 128 + (ot * 16 + hi * 8) * 2, c0);
            store_c_relu_bf16((char*)H1 + (1 * 16 + nloc) * 128 + (ot * 16 + hi * 8) * 2, c1);
        }

        // ============ layer 2: H2t = W2t[32x64] * H1t[64x32] (+b2, ReLU) ===
        Frag bh[2][2];
        #pragma unroll
        for (int nt = 0; nt < 2; ++nt)
            #pragma unroll
            for (int ks = 0; ks < 2; ++ks)
                bh[nt][ks] = load_bf((const char*)H1 + (nt * 16 + nloc) * 128 + ks * 64 + hi * 32);

        #pragma unroll
        for (int ot = 0; ot < 2; ++ot) {
            v8f c0, c1;
            #pragma unroll
            for (int r = 0; r < 8; ++r) {
                float bv = sB2[ot * 16 + hi * 8 + r];
                c0[r] = bv; c1[r] = bv;
            }
            #pragma unroll
            for (int ks = 0; ks < 2; ++ks) {
                Frag a = load_af((const char*)W2T + (ot * 16 + nloc) * 128 + ks * 64, hi);
                c0 = wmma_bf16(a.v, bh[0][ks].v, c0);
                c1 = wmma_bf16(a.v, bh[1][ks].v, c1);
            }
            // H2 row-major f32 [32 m][32 chan]; 8 consecutive channels/lane
            #pragma unroll
            for (int nt = 0; nt < 2; ++nt) {
                v8f c = nt ? c1 : c0;
                float* dst = (float*)((char*)H2 + (nt * 16 + nloc) * 128 + (ot * 16 + hi * 8) * 4);
                float4 o0 = { fmaxf(c[0], 0.f), fmaxf(c[1], 0.f), fmaxf(c[2], 0.f), fmaxf(c[3], 0.f) };
                float4 o1 = { fmaxf(c[4], 0.f), fmaxf(c[5], 0.f), fmaxf(c[6], 0.f), fmaxf(c[7], 0.f) };
                *(float4*)dst       = o0;
                *(float4*)(dst + 4) = o1;
            }
        }

        // ====== layer 3 (32->1) + sigmoid + mean over 20 neighbors =========
        __builtin_amdgcn_wave_barrier();
        float p = 0.f;
        if (lane < KNB) {
            float z = sB3[0];
            const float* hr = H2 + lane * 32;
            #pragma unroll
            for (int k = 0; k < 32; ++k) z += hr[k] * sW3[k];
            p = 1.f / (1.f + __expf(-z));
        }
        #pragma unroll
        for (int off = 16; off > 0; off >>= 1) p += __shfl_down(p, off);
        if (lane == 0) out[b] = p * (1.f / (float)KNB);
    }
}

extern "C" void kernel_launch(void* const* d_in, const int* in_sizes, int n_in,
                              void* d_out, int out_size, void* d_ws, size_t ws_size,
                              hipStream_t stream) {
    const int*   user_idxs       = (const int*)  d_in[0];
    const int*   item_idxs       = (const int*)  d_in[1];
    const int*   user_idx_tensor = (const int*)  d_in[2];
    const float* user_scr_tensor = (const float*)d_in[3];
    const int*   item_idx_tensor = (const int*)  d_in[4];
    const float* item_scr_tensor = (const float*)d_in[5];
    const float* user_emb_table  = (const float*)d_in[6];
    const float* item_emb_table  = (const float*)d_in[7];
    const float* w1 = (const float*)d_in[8];
    const float* b1 = (const float*)d_in[9];
    const float* w2 = (const float*)d_in[10];
    const float* b2 = (const float*)d_in[11];
    const float* w3 = (const float*)d_in[12];
    const float* b3 = (const float*)d_in[13];

    const int B = in_sizes[0];
    float* out = (float*)d_out;

    int blocks = 2048;
    if (blocks * WPB > B) blocks = (B + WPB - 1) / WPB;

    hipLaunchKernelGGL(cnn_fused_wmma_kernel, dim3(blocks), dim3(NTHR), 0, stream,
                       user_idxs, item_idxs,
                       user_idx_tensor, user_scr_tensor,
                       item_idx_tensor, item_scr_tensor,
                       user_emb_table, item_emb_table,
                       w1, b1, w2, b2, w3, b3, out, B);
}
